// WMSA_824633721572
// MI455X (gfx1250) — matrix operand order repaired and checked by reading, not compiled, not run on hardware
//
#include <hip/hip_runtime.h>
#include <stdint.h>

// ---------------- constants ----------------
#define BATCH   8
#define HIMG    256
#define WIMG    256
#define CCH     192
#define WS      8
#define SHIFT   4
#define NHEAD   6
#define HD      32
#define NPX     64            // tokens per window (8*8)
#define NWIN_Y  32            // HIMG/WS
#define NWIN_X  32
#define QKV_N   576           // 3*C
#define SCALE   0.17677669529663687f  // 32^-0.5

typedef __attribute__((ext_vector_type(16))) __bf16 v16bf;
typedef __attribute__((ext_vector_type(8)))  float  v8f;

union Frag16 {
    uint4 q[2];
    v16bf v;
};

__device__ __forceinline__ unsigned short f2bf(float f) {
    union { float f; unsigned u; } x; x.f = f;
    unsigned u = x.u;
    unsigned r = (u + 0x7FFFu + ((u >> 16) & 1u)) >> 16;  // RNE bf16
    return (unsigned short)r;
}
__device__ __forceinline__ unsigned pack2bf(float a, float b) {
    return (unsigned)f2bf(a) | ((unsigned)f2bf(b) << 16);
}

// A-matrix fragment (16x32 bf16), row-major [M][K] source, stride `ld`.
// ISA: lanes 0-15 = rows, K halves [0..7]&[16..23]; lanes 16-31: [8..15]&[24..31].
__device__ __forceinline__ v16bf load_frag_a(const unsigned short* base, int ld, int lane) {
    int row = lane & 15;
    int kh  = (lane >> 4) << 3;      // 0 or 8
    const unsigned short* p = base + row * ld + kh;
    Frag16 f;
    f.q[0] = *(const uint4*)(p);      // K kh..kh+7
    f.q[1] = *(const uint4*)(p + 16); // K 16+kh..23+kh
    return f.v;
}

// B-matrix fragment (32x16 bf16) from transposed [N][K] row-major source:
// element (k,n) = base[n*ld + k].
// ISA: lanes 0-15 = col n with K [0..7]&[16..23]; lanes 16-31: [8..15]&[24..31].
__device__ __forceinline__ v16bf load_frag_b(const unsigned short* base, int ld, int lane) {
    int n  = lane & 15;
    int kb = (lane >> 4) << 3;       // 0 or 8
    const unsigned short* p = base + n * ld + kb;
    Frag16 f;
    f.q[0] = *(const uint4*)(p);      // K kb..kb+7
    f.q[1] = *(const uint4*)(p + 16); // K 16+kb..23+kb
    return f.v;
}

__device__ __forceinline__ v8f wmma_bf16(v16bf a, v16bf b, v8f c) {
    return __builtin_amdgcn_wmma_f32_16x16x32_bf16(
        false, a, false, b, (short)0, c, false, false);
}

// ---------------- weight prep: transpose + f32->bf16 ----------------
__global__ __launch_bounds__(256)
void wmsa_prep_weights(const float* __restrict__ w_qkv,
                       const float* __restrict__ w_out,
                       unsigned short* __restrict__ wT_qkv,   // [576][192]
                       unsigned short* __restrict__ wT_out) { // [192][192]
    int idx = blockIdx.x * 256 + threadIdx.x;
    if (idx < QKV_N * CCH) {
        int n = idx / CCH, k = idx % CCH;
        wT_qkv[idx] = f2bf(w_qkv[k * QKV_N + n]);
    } else {
        int i2 = idx - QKV_N * CCH;
        if (i2 < CCH * CCH) {
            int n = i2 / CCH, k = i2 % CCH;
            wT_out[i2] = f2bf(w_out[k * CCH + n]);
        }
    }
}

// ---------------- fused W-MSA kernel: 1 block = 1 window ----------------
__global__ __launch_bounds__(256)
void wmsa_fused(const float* __restrict__ x,
                const float* __restrict__ b_qkv,
                const float* __restrict__ rel_pos,   // [6][15][15]
                const float* __restrict__ b_out,
                const unsigned short* __restrict__ wT_qkv,
                const unsigned short* __restrict__ wT_out,
                float* __restrict__ out) {
    // LDS: 24 + 72 + 16 + 8 = 120 KB -> 2 workgroups per WGP (320 KB cap)
    __shared__ __align__(16) unsigned short s_x[NPX * CCH];              // x bf16; reused as attn-out
    __shared__ __align__(16) unsigned short s_qkv[3 * NHEAD * NPX * HD]; // q,k row-major; v transposed
    __shared__ __align__(16) float          s_sim[NPX * NPX];            // sim[query][key]
    __shared__ __align__(16) unsigned short s_prob[NPX * NPX];           // probs bf16 [query][key]

    const int lane = threadIdx.x & 31;
    const int wave = threadIdx.x >> 5;
    const int col  = lane & 15;            // D-frag column
    const int half = (lane >> 4) << 3;     // D-frag row-base offset (0/8)

    const int blk = blockIdx.x;
    const int b   = blk >> 10;
    const int w   = blk & 1023;
    const int whi = w >> 5;
    const int wwi = w & 31;

    const float* xb = x + (size_t)b * HIMG * WIMG * CCH;

    // ---- Phase 0: load rolled x window -> LDS bf16 ----
    for (int e = threadIdx.x; e < NPX * (CCH / 4); e += 256) {
        int tok = e / (CCH / 4);
        int c4  = e % (CCH / 4);
        int ty = tok >> 3, tx = tok & 7;
        int sy = (whi * WS + ty + SHIFT) & (HIMG - 1);
        int sx = (wwi * WS + tx + SHIFT) & (WIMG - 1);
        float4 v = *(const float4*)(xb + ((size_t)sy * WIMG + sx) * CCH + c4 * 4);
        *(unsigned int*)(s_x + tok * CCH + c4 * 4 + 0) = pack2bf(v.x, v.y);
        *(unsigned int*)(s_x + tok * CCH + c4 * 4 + 2) = pack2bf(v.z, v.w);
    }
    __syncthreads();

    unsigned short* s_q = s_qkv;                        // [6][64][32] (tok-major)
    unsigned short* s_k = s_qkv + NHEAD * NPX * HD;     // [6][64][32]
    unsigned short* s_v = s_qkv + 2 * NHEAD * NPX * HD; // [6][32][64] (ch-major)

    // ---- Phase 1: QKV, computed transposed: D[ch][tok] = wT_qkv(A) x x(B) ----
    // M = 576 (36 tiles), N = 64 (4 tiles) -> 144 tiles over 8 waves, K = 6x32
    for (int t = wave; t < 144; t += 8) {
        int mt = t >> 2, nt = t & 3;
        v8f acc = {};
        for (int kk = 0; kk < 6; ++kk) {
            v16bf a  = load_frag_a(wT_qkv + (mt * 16) * CCH + kk * 32, CCH, lane);
            v16bf bf = load_frag_b(s_x + (nt * 16) * CCH + kk * 32, CCH, lane);
            acc = wmma_bf16(a, bf, acc);
        }
        int tok = nt * 16 + col;
        int m0  = mt * 16 + half;                 // global qkv channel base (8 consecutive)
        float4 bb0 = *(const float4*)(b_qkv + m0);
        float4 bb1 = *(const float4*)(b_qkv + m0 + 4);
        float vr[8] = { acc[0] + bb0.x, acc[1] + bb0.y, acc[2] + bb0.z, acc[3] + bb0.w,
                        acc[4] + bb1.x, acc[5] + bb1.y, acc[6] + bb1.z, acc[7] + bb1.w };
        // region select is tile-uniform (boundaries at multiples of 16 channels):
        // force it scalar so no exec-mask juggling appears in the hot loop.
        int region = __builtin_amdgcn_readfirstlane(mt) / 12;   // 0=q, 1=k, 2=v
        if (region == 0) {                        // q: contiguous [tok][ch] store
            int ch = m0;                          // 0..191
            unsigned short* dst = s_q + (ch >> 5) * (NPX * HD) + tok * HD + (ch & 31);
            uint4 pk = { pack2bf(vr[0], vr[1]), pack2bf(vr[2], vr[3]),
                         pack2bf(vr[4], vr[5]), pack2bf(vr[6], vr[7]) };
            *(uint4*)dst = pk;
        } else if (region == 1) {                 // k
            int ch = m0 - CCH;
            unsigned short* dst = s_k + (ch >> 5) * (NPX * HD) + tok * HD + (ch & 31);
            uint4 pk = { pack2bf(vr[0], vr[1]), pack2bf(vr[2], vr[3]),
                         pack2bf(vr[4], vr[5]), pack2bf(vr[6], vr[7]) };
            *(uint4*)dst = pk;
        } else {                                  // v: transposed [ch][tok]
            int ch = m0 - 2 * CCH;
            unsigned short* dst = s_v + (ch >> 5) * (NPX * HD) + (ch & 31) * NPX + tok;
            for (int r = 0; r < 8; ++r) dst[r * NPX] = f2bf(vr[r]);
        }
    }

    const bool edgeY = (whi == NWIN_Y - 1);
    const bool edgeX = (wwi == NWIN_X - 1);

    const int p_row = threadIdx.x >> 2;       // softmax: 4 lanes per query row
    const int sub   = threadIdx.x & 3;        // 16 keys each
    const int py = p_row >> 3, px = p_row & 7;

    // ---- Phase 2: attention per head ----
    for (int h = 0; h < NHEAD; ++h) {
        __syncthreads();

        const unsigned short* qh = s_q + h * (NPX * HD);
        const unsigned short* kh = s_k + h * (NPX * HD);
        const unsigned short* vh = s_v + h * (NPX * HD);

        // sim computed transposed: D[key][query] = k(A) x q(B); store sim[query][key]
        for (int i = 0; i < 2; ++i) {
            int t = wave * 2 + i;
            int mt = t >> 2, nt = t & 3;          // mt: key tile, nt: query tile
            v8f acc = {};
            v16bf a  = load_frag_a(kh + (mt * 16) * HD, HD, lane);
            v16bf bf = load_frag_b(qh + (nt * 16) * HD, HD, lane);
            acc = wmma_bf16(a, bf, acc);
            int qtok = nt * 16 + col;
            float* dst = s_sim + qtok * NPX + mt * 16 + half;   // 8 consecutive keys
            float4 lo = { acc[0] * SCALE, acc[1] * SCALE, acc[2] * SCALE, acc[3] * SCALE };
            float4 hi = { acc[4] * SCALE, acc[5] * SCALE, acc[6] * SCALE, acc[7] * SCALE };
            *(float4*)(dst)     = lo;
            *(float4*)(dst + 4) = hi;
        }
        __syncthreads();

        // softmax: 4 lanes per row, shfl-reduced; bias + shift-mask fused
        {
            const float* row = s_sim + p_row * NPX + sub * 16;
            const float* rp  = rel_pos + h * 225 + (py + 7) * 15 + (px + 7);
            float vals[16];
            float mx = -1e30f;
            for (int j = 0; j < 4; ++j) {
                float4 f4 = ((const float4*)row)[j];
                float fv[4] = { f4.x, f4.y, f4.z, f4.w };
                for (int e = 0; e < 4; ++e) {
                    int qi = sub * 16 + j * 4 + e;
                    int qy = qi >> 3, qx = qi & 7;
                    float s = fv[e] + rp[-(qy * 15 + qx)];
                    bool msk = (edgeY && ((py >= SHIFT) != (qy >= SHIFT))) ||
                               (edgeX && ((px >= SHIFT) != (qx >= SHIFT)));
                    s = msk ? -1e30f : s;
                    vals[j * 4 + e] = s;
                    mx = fmaxf(mx, s);
                }
            }
            mx = fmaxf(mx, __shfl_xor(mx, 1, 32));
            mx = fmaxf(mx, __shfl_xor(mx, 2, 32));
            float sum = 0.f;
            for (int e = 0; e < 16; ++e) {
                float ev = __expf(vals[e] - mx);
                vals[e] = ev;
                sum += ev;
            }
            sum += __shfl_xor(sum, 1, 32);
            sum += __shfl_xor(sum, 2, 32);
            float inv = 1.0f / sum;
            unsigned short* pdst = s_prob + p_row * NPX + sub * 16;
            uint4 pk0 = { pack2bf(vals[0] * inv,  vals[1] * inv),
                          pack2bf(vals[2] * inv,  vals[3] * inv),
                          pack2bf(vals[4] * inv,  vals[5] * inv),
                          pack2bf(vals[6] * inv,  vals[7] * inv) };
            uint4 pk1 = { pack2bf(vals[8] * inv,  vals[9] * inv),
                          pack2bf(vals[10] * inv, vals[11] * inv),
                          pack2bf(vals[12] * inv, vals[13] * inv),
                          pack2bf(vals[14] * inv, vals[15] * inv) };
            *(uint4*)(pdst)     = pk0;
            *(uint4*)(pdst + 8) = pk1;
        }
        __syncthreads();

        // PV computed transposed: D[ch][tok] = vT(A) x probs(B); store attn-out [tok][ch]
        {
            int mt = wave >> 2, nt = wave & 3;    // mt: channel tile (2), nt: query tile (4)
            v8f acc = {};
            for (int kk = 0; kk < 2; ++kk) {
                v16bf a  = load_frag_a(vh + (mt * 16) * NPX + kk * 32, NPX, lane);
                v16bf bf = load_frag_b(s_prob + (nt * 16) * NPX + kk * 32, NPX, lane);
                acc = wmma_bf16(a, bf, acc);
            }
            int tok = nt * 16 + col;
            int ch  = h * HD + mt * 16 + half;    // 8 consecutive channels
            unsigned short* dst = s_x + tok * CCH + ch;
            uint4 pk = { pack2bf(acc[0], acc[1]), pack2bf(acc[2], acc[3]),
                         pack2bf(acc[4], acc[5]), pack2bf(acc[6], acc[7]) };
            *(uint4*)dst = pk;
        }
    }
    __syncthreads();

    // ---- Phase 3: projection, transposed: D[ch][tok] = wT_out(A) x attnout(B) ----
    // M = 192 (12 tiles), N = 64 (4 tiles) -> 48 tiles over 8 waves, K = 6x32
    for (int t = wave; t < 48; t += 8) {
        int mt = t >> 2, nt = t & 3;
        v8f acc = {};
        for (int kk = 0; kk < 6; ++kk) {
            v16bf a  = load_frag_a(wT_out + (mt * 16) * CCH + kk * 32, CCH, lane);
            v16bf bf = load_frag_b(s_x + (nt * 16) * CCH + kk * 32, CCH, lane);
            acc = wmma_bf16(a, bf, acc);
        }
        int tok = nt * 16 + col;
        int ch0 = mt * 16 + half;                 // 8 consecutive output channels
        float4 bb0 = *(const float4*)(b_out + ch0);
        float4 bb1 = *(const float4*)(b_out + ch0 + 4);
        int ty = tok >> 3, tx = tok & 7;
        int sy = (whi * WS + ty + SHIFT) & (HIMG - 1);
        int sx = (wwi * WS + tx + SHIFT) & (WIMG - 1);
        float* dst = out + (((size_t)b * HIMG + sy) * WIMG + sx) * CCH + ch0;
        float4 lo = { acc[0] + bb0.x, acc[1] + bb0.y, acc[2] + bb0.z, acc[3] + bb0.w };
        float4 hi = { acc[4] + bb1.x, acc[5] + bb1.y, acc[6] + bb1.z, acc[7] + bb1.w };
        *(float4*)(dst)     = lo;
        *(float4*)(dst + 4) = hi;
    }
}

// ---------------- host launcher ----------------
extern "C" void kernel_launch(void* const* d_in, const int* in_sizes, int n_in,
                              void* d_out, int out_size, void* d_ws, size_t ws_size,
                              hipStream_t stream) {
    const float* x       = (const float*)d_in[0];
    const float* w_qkv   = (const float*)d_in[1];
    const float* b_qkv   = (const float*)d_in[2];
    const float* rel_pos = (const float*)d_in[3];
    const float* w_out   = (const float*)d_in[4];
    const float* b_out   = (const float*)d_in[5];
    float* out = (float*)d_out;

    unsigned short* wT_qkv = (unsigned short*)d_ws;   // 576*192 bf16
    unsigned short* wT_out = wT_qkv + QKV_N * CCH;    // 192*192 bf16

    int prep_elems = QKV_N * CCH + CCH * CCH;
    wmsa_prep_weights<<<(prep_elems + 255) / 256, 256, 0, stream>>>(
        w_qkv, w_out, wT_qkv, wT_out);

    int nblocks = BATCH * NWIN_Y * NWIN_X;            // 8192
    wmsa_fused<<<nblocks, 256, 0, stream>>>(
        x, b_qkv, rel_pos, b_out, wT_qkv, wT_out, out);
}
